// QuChater_43800076484691
// MI455X (gfx1250) — matrix operand order repaired and checked by reading, not compiled, not run on hardware
//
#include <hip/hip_runtime.h>
#include <hip/hip_bf16.h>

// ---------------------------------------------------------------------------
// QuChater hybrid TCN + LSTM + 6-qubit statevector model for MI455X (gfx1250)
//   - conv layers + LSTM input projection: f16 WMMA GEMMs (v_wmma_f32_16x16x32_f16)
//   - all shape params compile-time (no integer division in device code)
//   - K-pair-packed weights: B fragments fill each VGPR with one u32 load
//   - sequential scan: one wave32 per batch element; quantum state = 2 complex
//     amplitudes per lane, gates via __shfl_xor butterflies.
// ---------------------------------------------------------------------------

typedef __attribute__((ext_vector_type(16))) _Float16 v16h;
typedef __attribute__((ext_vector_type(2)))  _Float16 v2h;
typedef __attribute__((ext_vector_type(8)))  float    v8f;

#define BATCH 1024
#define TLEN  256
#define DIN   64
#define HID   32
#define BT    (BATCH * TLEN)

// ---------------- workspace layout (bytes) ----------------
#define OFF_XH    0L                                   // (BT,64) f16  = 32 MB
#define OFF_A1    (OFF_XH   + (long)BT*64*2)           // (BT,32) f16  = 16 MB
#define OFF_A2    (OFF_A1   + (long)BT*32*2)
#define OFF_A3    (OFF_A2   + (long)BT*32*2)
#define OFF_XPROJ (OFF_A3   + (long)BT*32*2)           // (BT,128) f32 = 128 MB
#define OFF_W0H   (OFF_XPROJ + (long)BT*128*4)
#define OFF_W1H   (OFF_W0H  + 3L*64*32*2)
#define OFF_W2H   (OFF_W1H  + 3L*32*32*2)
#define OFF_WIHH  (OFF_W2H  + 3L*32*32*2)
#define OFF_BSUM  (OFF_WIHH + 32L*128*2)

// ---------------- small prep kernels ----------------
__global__ void pack_f32_to_f16(const float* __restrict__ in,
                                _Float16* __restrict__ out, long n) {
  long i = (long)blockIdx.x * blockDim.x + threadIdx.x;
  if (i < n) out[i] = (_Float16)in[i];
}

// w (Cout,Cin,ntaps) f32 -> wpk K-pair-major f16:
//   wpk[tap*Cin*Cout + (ci>>1)*Cout*2 + co*2 + (ci&1)]
// so a (k,k+1) pair for one output column is a 4-byte-aligned u32.
__global__ void pack_conv_w(const float* __restrict__ w, _Float16* __restrict__ out,
                            int Cout, int Cin, int ntaps) {
  int i = blockIdx.x * blockDim.x + threadIdx.x;
  int total = Cout * Cin * ntaps;
  if (i >= total) return;
  int co = i % Cout;
  int rem = i / Cout;
  int ci = rem % Cin;
  int tap = rem / Cin;
  long o = (long)tap * Cin * Cout + (long)(ci >> 1) * Cout * 2 + co * 2 + (ci & 1);
  out[o] = (_Float16)w[(co * Cin + ci) * ntaps + tap];
}

__global__ void add_bias128(const float* __restrict__ a, const float* __restrict__ b,
                            float* __restrict__ out) {
  int i = threadIdx.x;
  if (i < 128) out[i] = a[i] + b[i];
}

// ---------------- WMMA fragment loaders (contiguous k = 16*(lane/16) + s in A and B) ----------------
// A: 16x32 f16, row-major in LDS, leading dim LD halves (LD even, K even -> u32).
template <int LD>
__device__ inline v16h load_Afrag(const _Float16* __restrict__ A) {
  int lane = threadIdx.x & 31;
  int m  = lane & 15;
  int kg = (lane >> 4) * 16;           // lanes 0-15: K 0-15; lanes 16-31: K 16-31
  v16h a;
#pragma unroll
  for (int i = 0; i < 8; ++i) {
    int k = kg + 2 * i;
    unsigned u = *(const unsigned*)(A + m * LD + k);
    v2h p = __builtin_bit_cast(v2h, u);
    a[2 * i]     = p[0];
    a[2 * i + 1] = p[1];
  }
  return a;
}

// B: 32x16 f16 sub-block from K-pair-packed weights, column offset n0.
template <int COUT>
__device__ inline v16h load_Bfrag(const _Float16* __restrict__ Bbase, int n0) {
  int lane = threadIdx.x & 31;
  int n = n0 + (lane & 15);
  int pg = (lane >> 4) * 8;            // pair-row base: lanes 0-15 K 0-15; 16-31 K 16-31
  v16h b;
#pragma unroll
  for (int i = 0; i < 8; ++i) {
    unsigned u = *(const unsigned*)(Bbase + (long)(pg + i) * COUT * 2 + n * 2);
    v2h p = __builtin_bit_cast(v2h, u);
    b[2 * i]     = p[0];
    b[2 * i + 1] = p[1];
  }
  return b;
}

// ---------------- fused conv-as-GEMM / projection kernel ----------------
// out[r, co] = act( bias[co] + sum_tap sum_ci in[r+tap-1, ci] * w[tap][ci][co] )
// rows shifted within each batch element's T range (zero padding).
template <int CIN, int COUT, int NTAPS, bool RELU, bool OUTH, bool OUTF>
__global__ void __launch_bounds__(128) gemm_conv_wmma(
    const _Float16* __restrict__ in, const _Float16* __restrict__ wpk,
    const float* __restrict__ bias, _Float16* __restrict__ outh,
    float* __restrict__ outf) {
  constexpr int LDA = CIN + 8;
  __shared__ _Float16 Asmem[4][16 * LDA];
  const int wave = threadIdx.x >> 5, lane = threadIdx.x & 31;
  const long tile = (long)blockIdx.x * 4 + wave;
  const long row0 = tile * 16;
  const int  t0   = (int)(row0 & (TLEN - 1));  // TLEN power of 2, tile within one b
  _Float16* As = Asmem[wave];

  v8f acc[COUT / 16];
#pragma unroll
  for (int n = 0; n < COUT / 16; ++n)
#pragma unroll
    for (int e = 0; e < 8; ++e) acc[n][e] = 0.0f;

#pragma unroll
  for (int tap = 0; tap < NTAPS; ++tap) {
    const int shift = (NTAPS == 3) ? (tap - 1) : 0;
    // stage (masked, shifted) 16 x CIN A tile into LDS, u32 granularity
#pragma unroll
    for (int it = 0; it < (16 * CIN / 2) / 32; ++it) {
      int idx = it * 32 + lane;
      int m  = idx / (CIN / 2);
      int kk = (idx % (CIN / 2)) * 2;
      int tt = t0 + m + shift;
      unsigned val = 0u;
      if (tt >= 0 && tt < TLEN)
        val = *(const unsigned*)(in + (row0 + m + shift) * CIN + kk);
      *(unsigned*)(As + m * LDA + kk) = val;
    }
    __syncthreads();
#pragma unroll
    for (int kc = 0; kc < CIN / 32; ++kc) {
      v16h a = load_Afrag<LDA>(As + kc * 32);
      const _Float16* Bbase = wpk + (long)(tap * CIN + kc * 32) * COUT;
#pragma unroll
      for (int n = 0; n < COUT / 16; ++n) {
        v16h b = load_Bfrag<COUT>(Bbase, n * 16);
        acc[n] = __builtin_amdgcn_wmma_f32_16x16x32_f16(
            false, a, false, b, (short)0, acc[n], false, false);
      }
    }
    __syncthreads();
  }

  const int nl = lane & 15, mb = (lane >> 4) * 8;
#pragma unroll
  for (int n = 0; n < COUT / 16; ++n) {
    int co = n * 16 + nl;
    float bv = bias ? bias[co] : 0.0f;
#pragma unroll
    for (int i = 0; i < 8; ++i) {
      int m = mb + i;
      float v = acc[n][i] + bv;
      if (RELU) v = fmaxf(v, 0.0f);
      long o = (row0 + m) * COUT + co;
      if constexpr (OUTH) outh[o] = (_Float16)v;
      if constexpr (OUTF) outf[o] = v;
    }
  }
}

// ---------------- quantum statevector helpers (wave32, 2 amps/lane) ----------
struct cplx { float x, y; };
__device__ inline cplx cmul(cplx a, cplx b) {
  return {a.x * b.x - a.y * b.y, a.x * b.y + a.y * b.x};
}
__device__ inline cplx cadd(cplx a, cplx b) { return {a.x + b.x, a.y + b.y}; }

// single-qubit gate on flat-index bit `bit` (bit 5 = amp-pair axis, local)
__device__ inline void apply1q(int lane, int bit, cplx g00, cplx g01, cplx g10,
                               cplx g11, cplx& s0, cplx& s1) {
  if (bit == 5) {
    cplx n0 = cadd(cmul(g00, s0), cmul(g01, s1));
    cplx n1 = cadd(cmul(g10, s0), cmul(g11, s1));
    s0 = n0; s1 = n1;
  } else {
    int mask = 1 << bit;
    bool up = (lane >> bit) & 1;
    cplx p0{__shfl_xor(s0.x, mask, 32), __shfl_xor(s0.y, mask, 32)};
    cplx p1{__shfl_xor(s1.x, mask, 32), __shfl_xor(s1.y, mask, 32)};
    cplx lo0 = up ? p0 : s0, hi0 = up ? s0 : p0;
    cplx lo1 = up ? p1 : s1, hi1 = up ? s1 : p1;
    s0 = up ? cadd(cmul(g10, lo0), cmul(g11, hi0))
            : cadd(cmul(g00, lo0), cmul(g01, hi0));
    s1 = up ? cadd(cmul(g10, lo1), cmul(g11, hi1))
            : cadd(cmul(g00, lo1), cmul(g01, hi1));
  }
}

// CNOT with control bit bc<5 and target bit bt<5
__device__ inline void cnot_gen(int lane, int bc, int bt, cplx& s0, cplx& s1) {
  int mask = 1 << bt;
  cplx p0{__shfl_xor(s0.x, mask, 32), __shfl_xor(s0.y, mask, 32)};
  cplx p1{__shfl_xor(s1.x, mask, 32), __shfl_xor(s1.y, mask, 32)};
  if ((lane >> bc) & 1) { s0 = p0; s1 = p1; }
}

__device__ inline float sigm(float x) { return 1.0f / (1.0f + expf(-x)); }

// ---------------- sequential scan: one wave32 per batch element -------------
__global__ void __launch_bounds__(128) scan_kernel(
    const float* __restrict__ Xproj,  // (B*T, 128) = xt@Wih^T + bih + bhh
    const float* __restrict__ Whh,    // (128, 32)
    const float* __restrict__ qw,     // (3, 6)
    const float* __restrict__ Wfc,    // (32, 6)
    const float* __restrict__ bfc,    // (32)
    const float* __restrict__ Wout,   // (1, 32)
    const float* __restrict__ bout,   // (1)
    float* __restrict__ out) {
  __shared__ float Wl[128 * 32];
  for (int i = threadIdx.x; i < 128 * 32; i += blockDim.x) Wl[i] = Whh[i];
  __syncthreads();

  const int wave = threadIdx.x >> 5, lane = threadIdx.x & 31;
  const int b = blockIdx.x * 4 + wave;
  const float* xp = Xproj + (long)b * TLEN * 128;

  float wfc_l[6];
#pragma unroll
  for (int i = 0; i < 6; ++i) wfc_l[i] = Wfc[lane * 6 + i];
  const float bfc_l = bfc[lane];
  const float wout_l = Wout[lane];

  float qc[18], qs[18];
#pragma unroll
  for (int g = 0; g < 18; ++g) {
    float a = 0.5f * qw[g];
    qc[g] = cosf(a); qs[g] = sinf(a);
  }

  float h = 0.0f, c = 0.0f;
  for (int t = 0; t < TLEN; ++t) {
    const float* gp = xp + (long)t * 128;
    float gi = gp[lane], gf = gp[32 + lane], gg = gp[64 + lane], go = gp[96 + lane];
    if (t + 1 < TLEN) __builtin_prefetch(gp + 128 + lane, 0, 0);  // global_prefetch_b8

    // recurrent matvec h @ Whh^T, Whh staged in LDS
#pragma unroll
    for (int k = 0; k < 32; ++k) {
      float hk = __shfl(h, k, 32);
      gi = fmaf(hk, Wl[lane * 32 + k], gi);
      gf = fmaf(hk, Wl[(lane + 32) * 32 + k], gf);
      gg = fmaf(hk, Wl[(lane + 64) * 32 + k], gg);
      go = fmaf(hk, Wl[(lane + 96) * 32 + k], go);
    }
    c = sigm(gf) * c + sigm(gi) * tanhf(gg);
    h = sigm(go) * tanhf(c);

    // logistic map
    h = 3.99f * h * (1.0f - h);
    // Henon map on components 0,1
    float x0 = __shfl(h, 0, 32), y0 = __shfl(h, 1, 32);
    if (lane == 0)      h = 1.0f - 1.4f * x0 * x0 + y0;
    else if (lane == 1) h = 0.3f * x0;

    // ---- 6-qubit statevector sim; qubit q <-> flat-index bit (5-q) ----
    float ang[6];
#pragma unroll
    for (int q = 0; q < 6; ++q) ang[q] = 0.5f * __shfl(h, q, 32);

    cplx s0{lane == 0 ? 1.0f : 0.0f, 0.0f}, s1{0.0f, 0.0f};
#pragma unroll
    for (int i = 0; i < 6; ++i) {
      const int bit = 5 - i;
      float ax = ang[i], ay = ang[(i + 1) % 6], az = ang[(i + 2) % 6];
      float cx = cosf(ax), sx = sinf(ax);
      apply1q(lane, bit, {cx, 0.f}, {0.f, -sx}, {0.f, -sx}, {cx, 0.f}, s0, s1);
      float cy = cosf(ay), sy = sinf(ay);
      apply1q(lane, bit, {cy, 0.f}, {-sy, 0.f}, {sy, 0.f}, {cy, 0.f}, s0, s1);
      float cz = cosf(az), sz = sinf(az);
      apply1q(lane, bit, {cz, -sz}, {0.f, 0.f}, {0.f, 0.f}, {cz, sz}, s0, s1);
    }
#pragma unroll
    for (int l = 0; l < 3; ++l) {
#pragma unroll
      for (int q = 0; q < 6; ++q) {
        float cg = qc[l * 6 + q], sg = qs[l * 6 + q];
        apply1q(lane, 5 - q, {cg, 0.f}, {-sg, 0.f}, {sg, 0.f}, {cg, 0.f}, s0, s1);
      }
      // CNOT ring (q, q+1 mod 6): bits (5,4),(4,3),(3,2),(2,1),(1,0),(0,5)
      s1.x = __shfl_xor(s1.x, 16, 32);   // (0,1): control = amp axis, flip bit 4 in s1
      s1.y = __shfl_xor(s1.y, 16, 32);
      cnot_gen(lane, 4, 3, s0, s1);
      cnot_gen(lane, 3, 2, s0, s1);
      cnot_gen(lane, 2, 1, s0, s1);
      cnot_gen(lane, 1, 0, s0, s1);
      if (lane & 1) { cplx tmp = s0; s0 = s1; s1 = tmp; }  // (5,0): target = amp axis
    }

    // expectation values <Z_i>
    float p0 = s0.x * s0.x + s0.y * s0.y;
    float p1 = s1.x * s1.x + s1.y * s1.y;
    float ev[6];
    ev[0] = p0 - p1;                    // bit 5 = amp axis
#pragma unroll
    for (int i = 1; i < 6; ++i) {
      int bit = 5 - i;
      float sgn = ((lane >> bit) & 1) ? -1.0f : 1.0f;
      ev[i] = sgn * (p0 + p1);
    }
#pragma unroll
    for (int i = 0; i < 6; ++i)
#pragma unroll
      for (int off = 16; off >= 1; off >>= 1)
        ev[i] += __shfl_xor(ev[i], off, 32);

    float a = bfc_l;
#pragma unroll
    for (int i = 0; i < 6; ++i) a = fmaf(ev[i], wfc_l[i], a);
    h += a;
  }

  // out = sigmoid(h @ Wout^T + bout)
  float v = h * wout_l;
#pragma unroll
  for (int off = 16; off >= 1; off >>= 1) v += __shfl_xor(v, off, 32);
  if (lane == 0) out[b] = 1.0f / (1.0f + expf(-(v + bout[0])));
}

// ---------------------------------------------------------------------------
extern "C" void kernel_launch(void* const* d_in, const int* in_sizes, int n_in,
                              void* d_out, int out_size, void* d_ws, size_t ws_size,
                              hipStream_t stream) {
  const float* x    = (const float*)d_in[0];
  const float* w0   = (const float*)d_in[1];
  const float* b0   = (const float*)d_in[2];
  const float* w1   = (const float*)d_in[3];
  const float* b1   = (const float*)d_in[4];
  const float* w2   = (const float*)d_in[5];
  const float* b2   = (const float*)d_in[6];
  const float* Wih  = (const float*)d_in[7];
  const float* Whh  = (const float*)d_in[8];
  const float* bih  = (const float*)d_in[9];
  const float* bhh  = (const float*)d_in[10];
  const float* qwts = (const float*)d_in[11];
  const float* Wfc  = (const float*)d_in[12];
  const float* bfc  = (const float*)d_in[13];
  const float* Wout = (const float*)d_in[14];
  const float* bout = (const float*)d_in[15];
  (void)in_sizes; (void)n_in; (void)out_size; (void)ws_size;

  char* ws = (char*)d_ws;
  _Float16* xh    = (_Float16*)(ws + OFF_XH);
  _Float16* a1    = (_Float16*)(ws + OFF_A1);
  _Float16* a2    = (_Float16*)(ws + OFF_A2);
  _Float16* a3    = (_Float16*)(ws + OFF_A3);
  float*    Xproj = (float*)   (ws + OFF_XPROJ);
  _Float16* w0h   = (_Float16*)(ws + OFF_W0H);
  _Float16* w1h   = (_Float16*)(ws + OFF_W1H);
  _Float16* w2h   = (_Float16*)(ws + OFF_W2H);
  _Float16* wihh  = (_Float16*)(ws + OFF_WIHH);
  float*    bsum  = (float*)   (ws + OFF_BSUM);

  // prep: f16 packing of x and weights (K-pair-major B layout), fused LSTM bias
  long nx = (long)BT * DIN;
  pack_f32_to_f16<<<(int)((nx + 255) / 256), 256, 0, stream>>>(x, xh, nx);
  pack_conv_w<<<(32 * 64 * 3 + 255) / 256, 256, 0, stream>>>(w0, w0h, 32, 64, 3);
  pack_conv_w<<<(32 * 32 * 3 + 255) / 256, 256, 0, stream>>>(w1, w1h, 32, 32, 3);
  pack_conv_w<<<(32 * 32 * 3 + 255) / 256, 256, 0, stream>>>(w2, w2h, 32, 32, 3);
  pack_conv_w<<<(128 * 32 + 255) / 256, 256, 0, stream>>>(Wih, wihh, 128, 32, 1);
  add_bias128<<<1, 128, 0, stream>>>(bih, bhh, bsum);

  // TCN conv stack (WMMA GEMMs, ReLU epilogue)
  gemm_conv_wmma<64, 32, 3, true, true, false>
      <<<BT / 64, 128, 0, stream>>>(xh, w0h, b0, a1, nullptr);
  gemm_conv_wmma<32, 32, 3, true, true, false>
      <<<BT / 64, 128, 0, stream>>>(a1, w1h, b1, a2, nullptr);
  gemm_conv_wmma<32, 32, 3, true, true, false>
      <<<BT / 64, 128, 0, stream>>>(a2, w2h, b2, a3, nullptr);

  // LSTM input projection Xproj = a3 @ Wih^T + (bih + bhh)  (WMMA GEMM)
  gemm_conv_wmma<32, 128, 1, false, false, true>
      <<<BT / 64, 128, 0, stream>>>(a3, wihh, bsum, nullptr, Xproj);

  // sequential scan: 1024 waves, 4 waves/block
  scan_kernel<<<BATCH / 4, 128, 0, stream>>>(Xproj, Whh, qwts, Wfc, bfc, Wout,
                                             bout, (float*)d_out);
}
